// DKT2_3487513444374
// MI455X (gfx1250) — matrix operand order repaired and checked by reading, not compiled, not run on hardware
//
#include <hip/hip_runtime.h>
#include <cstdint>

// ---------------------------------------------------------------- types
typedef __bf16 v16bf __attribute__((ext_vector_type(16)));
typedef float  v8f   __attribute__((ext_vector_type(8)));

__device__ __forceinline__ __bf16 f2bf(float f) {
  unsigned u = __builtin_bit_cast(unsigned, f);
  unsigned r = (u + 0x7FFFu + ((u >> 16) & 1u)) >> 16;
  unsigned short s = (unsigned short)r;
  return __builtin_bit_cast(__bf16, s);
}
__device__ __forceinline__ float logsigf(float x) {
  return fminf(x, 0.f) - log1pf(expf(-fabsf(x)));
}
__device__ __forceinline__ float siluf(float x) { return x / (1.f + expf(-x)); }
__device__ __forceinline__ float sigmf(float x) { return 1.f / (1.f + expf(-x)); }

// Model constants
#define TOK   16384      // B*S
#define SEQ   512
#define NB    32
#define EDIM  320
#define IDIM  416
#define DHM   208
#define DHS   160
#define NHX   2

// ================================================================ GEMM
// C[M,N] (+=) act( A[M,K] * B[K,N] + bias ), bf16 WMMA, f32 accum.
// Workgroup tile 128x64; 8 waves; wave w -> M-subtile w, loops 4 N-subtiles.
// A tile: async global->LDS (f32, double-buffered, GLOBAL_LOAD_ASYNC_TO_LDS_B128).
// B tile: register-prefetched, stored transposed as bf16 (double-buffered).
#define A_STR 36   // f32 row stride (16B-aligned rows for B128 async copy)
#define B_STR 34   // bf16 row stride (17 dwords, odd -> conflict-free pairs)

__global__ __launch_bounds__(256) void k_gemm(
    const float* __restrict__ A, const float* __restrict__ Bw,
    const float* __restrict__ bias, float* __restrict__ C,
    int M, int N, int K, int lda, int ldb, int ldc, int accum, int act)
{
  __shared__ float  Asf[2 * 128 * A_STR];  // 36,864 B
  __shared__ __bf16 Bs[2 * 64 * B_STR];    //  8,704 B
  const int tid = threadIdx.x;
  const int wid = tid >> 5, lane = tid & 31;
  const int hf = lane >> 4, mn = lane & 15;
  const int m0 = blockIdx.x * 128, n0 = blockIdx.y * 64;

  const unsigned long long Abase = (unsigned long long)A;
  const unsigned ldsA0 = (unsigned)(uintptr_t)(&Asf[0]);

  v8f acc[4] = {};
  float breg[8];

  // issue async copy of A chunk kc into ring slot `buf`
  auto issueA = [&](int kc, int buf) {
    const int k0 = kc * 32;
    const unsigned lbase = ldsA0 + (unsigned)(buf * 128 * A_STR * 4);
    #pragma unroll
    for (int p = 0; p < 4; ++p) {
      int id = p * 256 + tid;          // 1024 B128 lane-ops: 128 rows x 8 quads
      int row = id >> 3, c4 = id & 7;
      unsigned lds = lbase + (unsigned)((row * A_STR + c4 * 4) * 4);
      unsigned gof = (unsigned)(((size_t)(m0 + row) * lda + k0 + c4 * 4) * 4);
      asm volatile("global_load_async_to_lds_b128 %0, %1, %2 offset:0"
                   :: "v"(lds), "v"(gof), "s"(Abase) : "memory");
    }
  };
  // prefetch B chunk kc into registers
  auto loadB = [&](int kc) {
    const int k0 = kc * 32;
    #pragma unroll
    for (int p = 0; p < 8; ++p) {
      int idx = p * 256 + tid;
      int kk = idx >> 6, nn = idx & 63;
      breg[p] = (n0 + nn < N) ? Bw[(size_t)(k0 + kk) * ldb + n0 + nn] : 0.f;
    }
  };
  auto storeB = [&](int buf) {
    __bf16* Bb = Bs + buf * 64 * B_STR;
    #pragma unroll
    for (int p = 0; p < 8; ++p) {
      int idx = p * 256 + tid;
      int kk = idx >> 6, nn = idx & 63;
      Bb[nn * B_STR + kk] = f2bf(breg[p]);
    }
  };

  const int nchunk = K >> 5;
  issueA(0, 0);
  loadB(0);
  for (int kc = 0; kc < nchunk; ++kc) {
    const int buf = kc & 1;
    asm volatile("s_wait_asynccnt 0x0" ::: "memory");  // own A-copy done
    storeB(buf);
    __syncthreads();                                    // all copies visible
    if (kc + 1 < nchunk) { issueA(kc + 1, buf ^ 1); loadB(kc + 1); }

    const float*  Ab = Asf + buf * 128 * A_STR;
    const __bf16* Bb = Bs  + buf * 64 * B_STR;
    // A fragment (documented 16-bit A layout), cvt co-executes with WMMA
    v16bf af;
    const int arow = wid * 16 + mn;
    #pragma unroll
    for (int j = 0; j < 8; ++j) {
      int kk = ((j >> 2) << 4) + (hf << 3) + ((j & 3) << 1);
      af[2 * j]     = f2bf(Ab[arow * A_STR + kk]);
      af[2 * j + 1] = f2bf(Ab[arow * A_STR + kk + 1]);
    }
    #pragma unroll
    for (int nt = 0; nt < 4; ++nt) {
      v16bf bfr;
      const int brow = nt * 16 + mn;
      #pragma unroll
      for (int j = 0; j < 8; ++j) {
        int kk = ((j >> 2) << 4) + (hf << 3) + ((j & 3) << 1);
        bfr[2 * j]     = Bb[brow * B_STR + kk];
        bfr[2 * j + 1] = Bb[brow * B_STR + kk + 1];
      }
      acc[nt] = __builtin_amdgcn_wmma_f32_16x16x32_bf16(
          false, af, false, bfr, (short)0, acc[nt], false, false);
    }
  }
  // epilogue (C layout: VGPR r -> M = r + 8*half, N = lane&15)
  const int hi8 = hf << 3;
  for (int nt = 0; nt < 4; ++nt) {
    int col = n0 + nt * 16 + mn;
    if (col >= N) continue;
    float bv = bias ? bias[col] : 0.f;
    #pragma unroll
    for (int r = 0; r < 8; ++r) {
      int row = m0 + wid * 16 + r + hi8;
      size_t off = (size_t)row * ldc + col;
      float v = acc[nt][r] + bv;
      if (accum) v += C[off];
      if (act == 1) v = fmaxf(v, 0.f);
      else if (act == 2) v = sigmf(v);
      C[off] = v;
    }
  }
}

// ================================================================ embed
__global__ void k_embed(const int* __restrict__ cin, const int* __restrict__ rin,
                        const float* __restrict__ qemb, const float* __restrict__ qaemb,
                        const float* __restrict__ qdiff, const float* __restrict__ qadiff,
                        const float* __restrict__ dif,
                        float* __restrict__ x, float* __restrict__ qeo, float* __restrict__ pidb)
{
  int tok = blockIdx.x, e = threadIdx.x;
  int c = min(max(cin[tok], 0), 999);
  int rr = min(max(rin[tok], 0), 1);
  float pid = dif[c];
  float qe = qemb[c * EDIM + e];
  x[(size_t)tok * EDIM + e] = qaemb[rr * EDIM + e] + qe + pid * qadiff[(size_t)(rr + 2 * c) * EDIM + e];
  qeo[(size_t)tok * EDIM + e] = qe + pid * qdiff[c * EDIM + e];
  if (e == 0) pidb[tok] = pid;
}

// ================================================================ layernorm (weight-only), wave per token
__global__ __launch_bounds__(256) void k_ln(const float* __restrict__ in,
                                            const float* __restrict__ w,
                                            float* __restrict__ out, int nan2num)
{
  int tok = blockIdx.x * 8 + (threadIdx.x >> 5);
  int lane = threadIdx.x & 31;
  const float* xp = in + (size_t)tok * EDIM;
  float vals[10], s1 = 0.f, s2 = 0.f;
  #pragma unroll
  for (int i = 0; i < 10; ++i) {
    float v = xp[lane + 32 * i];
    vals[i] = v; s1 += v; s2 += v * v;
  }
  #pragma unroll
  for (int off = 16; off; off >>= 1) {
    s1 += __shfl_xor(s1, off, 32);
    s2 += __shfl_xor(s2, off, 32);
  }
  float mu = s1 * (1.f / EDIM);
  float var = s2 * (1.f / EDIM) - mu * mu;
  float rs = rsqrtf(var + 1e-5f);
  #pragma unroll
  for (int i = 0; i < 10; ++i) {
    float o = (vals[i] - mu) * rs * w[lane + 32 * i];
    if (nan2num) {
      if (o != o) o = 0.f;
      else if (o > 3.402823466e38f) o = 1.f;
      else if (o < -3.402823466e38f) o = -1.f;
    }
    out[(size_t)tok * EDIM + lane + 32 * i] = o;
  }
}

// ================================================================ depthwise causal conv (K=4) + SiLU
__global__ void k_conv_silu(const float* __restrict__ in, int ldin,
                            const float* __restrict__ w, const float* __restrict__ bias,
                            float* __restrict__ out, int ldout, int C)
{
  int idx = blockIdx.x * 256 + threadIdx.x;
  if (idx >= TOK * C) return;
  int c = idx % C, tok = idx / C;
  int s = tok & (SEQ - 1), b = tok >> 9;
  float acc = bias[c];
  #pragma unroll
  for (int kk = 0; kk < 4; ++kk) {
    int sp = s - 3 + kk;
    if (sp >= 0) acc += in[(size_t)(b * SEQ + sp) * ldin + c] * w[kk * C + c];
  }
  out[(size_t)tok * ldout + c] = siluf(acc);
}

// ================================================================ block-diagonal headwise linear (104 blocks of 4x4)
__global__ void k_headwise(const float* __restrict__ in, int ldin,
                           const float* __restrict__ w, const float* __restrict__ bias,
                           float* __restrict__ out)
{
  int idx = blockIdx.x * 256 + threadIdx.x;
  if (idx >= TOK * 104) return;
  int blk = idx % 104, tok = idx / 104;
  const float* xp = in + (size_t)tok * ldin + blk * 4;
  const float* wp = w + blk * 16;
  float x0 = xp[0], x1 = xp[1], x2 = xp[2], x3 = xp[3];
  #pragma unroll
  for (int o = 0; o < 4; ++o) {
    float y = wp[o * 4 + 0] * x0 + wp[o * 4 + 1] * x1 +
              wp[o * 4 + 2] * x2 + wp[o * 4 + 3] * x3 + bias[blk * 4 + o];
    out[(size_t)tok * IDIM + blk * 4 + o] = y;
  }
}

// ================================================================ ig/fg skinny projections (1248 -> 2), wave per token
__global__ __launch_bounds__(256) void k_igfg(
    const float* __restrict__ q, const float* __restrict__ k, const float* __restrict__ v,
    const float* __restrict__ igw, const float* __restrict__ igb,
    const float* __restrict__ fgw, const float* __restrict__ fgb,
    float* __restrict__ ig, float* __restrict__ fg)
{
  int tok = blockIdx.x * 8 + (threadIdx.x >> 5);
  int lane = threadIdx.x & 31;
  float i0 = 0.f, i1 = 0.f, f0 = 0.f, f1 = 0.f;
  for (int j = lane; j < 3 * IDIM; j += 32) {
    float g;
    if (j < IDIM) g = q[(size_t)tok * IDIM + j];
    else if (j < 2 * IDIM) g = k[(size_t)tok * IDIM + j - IDIM];
    else g = v[(size_t)tok * IDIM + j - 2 * IDIM];
    i0 += g * igw[j * 2];  i1 += g * igw[j * 2 + 1];
    f0 += g * fgw[j * 2];  f1 += g * fgw[j * 2 + 1];
  }
  #pragma unroll
  for (int off = 16; off; off >>= 1) {
    i0 += __shfl_xor(i0, off, 32); i1 += __shfl_xor(i1, off, 32);
    f0 += __shfl_xor(f0, off, 32); f1 += __shfl_xor(f1, off, 32);
  }
  if (lane == 0) {
    int b = tok >> 9, s = tok & (SEQ - 1);
    ig[(size_t)(b * NHX + 0) * SEQ + s] = i0 + igb[0];
    ig[(size_t)(b * NHX + 1) * SEQ + s] = i1 + igb[1];
    fg[(size_t)(b * NHX + 0) * SEQ + s] = f0 + fgb[0];
    fg[(size_t)(b * NHX + 1) * SEQ + s] = f1 + fgb[1];
  }
}

// ================================================================ mLSTM stabilizer scan: alpha, prefix-max, exp(-maxd)
__global__ void k_mscan(const float* __restrict__ ig, const float* __restrict__ fg,
                        float* __restrict__ alpha, float* __restrict__ mrow,
                        float* __restrict__ en)
{
  int bh = threadIdx.x;
  if (bh >= NB * NHX) return;
  size_t base = (size_t)bh * SEQ;
  float lfc = 0.f, mr = -1e30f;
  for (int t = 0; t < SEQ; ++t) {
    lfc += logsigf(fg[base + t]);
    float a = ig[base + t] - lfc;
    mr = fmaxf(mr, a);
    alpha[base + t] = a;
    mrow[base + t] = mr;
    en[base + t] = expf(-(lfc + mr));
  }
}

// ================================================================ fused mLSTM attention: per (b, head, 64-query block)
__global__ __launch_bounds__(256) void k_attn(
    const float* __restrict__ q, const float* __restrict__ kmat, const float* __restrict__ vmat,
    const float* __restrict__ alpha, const float* __restrict__ mrow, const float* __restrict__ en,
    float* __restrict__ hout)
{
  extern __shared__ char smem_dyn[];
  float* scores = (float*)smem_dyn;        // 64 x 521 (odd-dword padded)
  float* al  = scores + 64 * 521;          // 512
  float* mrl = al + 512;                   // 64
  float* enl = mrl + 64;                   // 64
  float* red = enl + 64;                   // 64 x 4
  float* scl = red + 256;                  // 64

  const int tid = threadIdx.x, wid = tid >> 5, lane = tid & 31;
  const int hf = lane >> 4, mn = lane & 15, hi8 = hf << 3;
  const int s0 = blockIdx.x * 64;
  const int hh = blockIdx.y, b = blockIdx.z;
  const size_t bh = (size_t)(b * NHX + hh) * SEQ;

  for (int i = tid; i < SEQ; i += 256) al[i] = alpha[bh + i];
  if (tid < 64) { mrl[tid] = mrow[bh + s0 + tid]; enl[tid] = en[bh + s0 + tid]; }
  __syncthreads();

  const float isq = 0.0693375245f;  // 1/sqrt(208)
  const int ssub = wid & 3, tpart = wid >> 2;

  // hoist Q fragments (7 K-chunks covering DHM=208, zero-padded to 224)
  v16bf qf[7];
  {
    const int srow = s0 + ssub * 16 + mn;
    const size_t qbase = (size_t)(b * SEQ + srow) * IDIM + hh * DHM;
    for (int kc = 0; kc < 7; ++kc) {
      #pragma unroll
      for (int j = 0; j < 8; ++j) {
        int d = kc * 32 + ((j >> 2) << 4) + (hf << 3) + ((j & 3) << 1);
        qf[kc][2 * j]     = (d < DHM)     ? f2bf(q[qbase + d])     : f2bf(0.f);
        qf[kc][2 * j + 1] = (d + 1 < DHM) ? f2bf(q[qbase + d + 1]) : f2bf(0.f);
      }
    }
  }

  // phase 1: scores = causal-masked (QK^T/sqrt(d)) * exp(alpha[t]-mr[s])
  const int smin = s0 + ssub * 16;
  for (int i = 0; i < 16; ++i) {
    const int t0 = (tpart * 16 + i) * 16;
    if (t0 > smin + 15) {  // fully masked tile -> zeros
      #pragma unroll
      for (int r = 0; r < 8; ++r)
        scores[(ssub * 16 + r + hi8) * 521 + t0 + mn] = 0.f;
      continue;
    }
    v8f acc = {};
    const size_t kbase = (size_t)(b * SEQ + t0 + mn) * IDIM + hh * DHM;
    for (int kc = 0; kc < 7; ++kc) {
      v16bf kf;
      #pragma unroll
      for (int j = 0; j < 8; ++j) {
        int d = kc * 32 + ((j >> 2) << 4) + (hf << 3) + ((j & 3) << 1);
        kf[2 * j]     = (d < DHM)     ? f2bf(kmat[kbase + d])     : f2bf(0.f);
        kf[2 * j + 1] = (d + 1 < DHM) ? f2bf(kmat[kbase + d + 1]) : f2bf(0.f);
      }
      acc = __builtin_amdgcn_wmma_f32_16x16x32_bf16(
          false, qf[kc], false, kf, (short)0, acc, false, false);
    }
    const int t = t0 + mn;
    #pragma unroll
    for (int r = 0; r < 8; ++r) {
      int sl = ssub * 16 + r + hi8;
      float val = 0.f;
      if (t <= s0 + sl) val = acc[r] * isq * expf(al[t] - mrl[sl]);
      scores[sl * 521 + t] = val;
    }
  }
  __syncthreads();

  // phase 2: normalize rows: / (max(|rowsum|, exp(-maxd)) + 1e-6)
  {
    int r = tid >> 2, qp = tid & 3;
    float ssum = 0.f;
    for (int t = qp * 128; t < qp * 128 + 128; ++t) ssum += scores[r * 521 + t];
    red[r * 4 + qp] = ssum;
  }
  __syncthreads();
  if (tid < 64) {
    float rs = red[tid * 4] + red[tid * 4 + 1] + red[tid * 4 + 2] + red[tid * 4 + 3];
    scl[tid] = 1.f / (fmaxf(fabsf(rs), enl[tid]) + 1e-6f);
  }
  __syncthreads();
  for (int i = tid; i < 64 * SEQ; i += 256) {
    int rr = i >> 9, t = i & (SEQ - 1);
    scores[rr * 521 + t] *= scl[rr];
  }
  __syncthreads();

  // phase 3: out = scores @ V, K truncated at causal limit
  const int nkc = (s0 + 64) >> 5;
  for (int tt = wid; tt < 52; tt += 8) {
    const int ss = tt & 3, d0 = (tt >> 2) * 16;
    v8f acc = {};
    const int arow = ss * 16 + mn;
    for (int kc = 0; kc < nkc; ++kc) {
      const int t0 = kc * 32;
      v16bf af, bfr;
      #pragma unroll
      for (int j = 0; j < 8; ++j) {
        int tk = t0 + ((j >> 2) << 4) + (hf << 3) + ((j & 3) << 1);
        af[2 * j]     = f2bf(scores[arow * 521 + tk]);
        af[2 * j + 1] = f2bf(scores[arow * 521 + tk + 1]);
      }
      #pragma unroll
      for (int j = 0; j < 8; ++j) {
        int tk = t0 + ((j >> 2) << 4) + (hf << 3) + ((j & 3) << 1);
        size_t v0 = (size_t)(b * SEQ + tk) * IDIM + hh * DHM + d0 + mn;
        bfr[2 * j]     = f2bf(vmat[v0]);
        bfr[2 * j + 1] = f2bf(vmat[v0 + IDIM]);
      }
      acc = __builtin_amdgcn_wmma_f32_16x16x32_bf16(
          false, af, false, bfr, (short)0, acc, false, false);
    }
    #pragma unroll
    for (int r = 0; r < 8; ++r) {
      int s = s0 + ss * 16 + r + hi8;
      hout[(size_t)(b * SEQ + s) * IDIM + hh * DHM + d0 + mn] = acc[r];
    }
  }
}

// ================================================================ mLSTM multi-head norm + skip + SiLU(z) gate
__global__ __launch_bounds__(256) void k_mhnorm_gate(
    const float* __restrict__ hbuf, const float* __restrict__ xc,
    const float* __restrict__ xi, const float* __restrict__ outnorm,
    const float* __restrict__ skip, float* __restrict__ outb)
{
  int gid = blockIdx.x * 8 + (threadIdx.x >> 5);  // token*2 + head
  int lane = threadIdx.x & 31;
  int hh = gid & 1, tok = gid >> 1;
  size_t base = (size_t)tok * IDIM + hh * DHM;
  float vals[7], s1 = 0.f, s2 = 0.f;
  #pragma unroll
  for (int i = 0; i < 7; ++i) {
    int d = lane + 32 * i;
    float v = (d < DHM) ? hbuf[base + d] : 0.f;
    vals[i] = v; s1 += v; s2 += v * v;
  }
  #pragma unroll
  for (int off = 16; off; off >>= 1) {
    s1 += __shfl_xor(s1, off, 32);
    s2 += __shfl_xor(s2, off, 32);
  }
  float mu = s1 * (1.f / DHM);
  float var = s2 * (1.f / DHM) - mu * mu;
  float rs = rsqrtf(var + 1e-5f);
  #pragma unroll
  for (int i = 0; i < 7; ++i) {
    int d = lane + 32 * i;
    if (d >= DHM) break;
    int cch = hh * DHM + d;
    float hn = (vals[i] - mu) * rs * outnorm[cch];
    float z = xi[(size_t)tok * 832 + IDIM + cch];
    outb[(size_t)tok * IDIM + cch] = (hn + skip[cch] * xc[(size_t)tok * IDIM + cch]) * siluf(z);
  }
}

// ================================================================ sLSTM recurrent scan: one WG per (b,h); Rg bf16 in LDS
__global__ __launch_bounds__(256) void k_sscan(
    const float* __restrict__ Rg, const float* __restrict__ pre, float* __restrict__ ys)
{
  extern __shared__ char smem_dyn[];
  __bf16* R = (__bf16*)smem_dyn;                  // [4][160][166] transposed [g][e][d]
  float* ylds = (float*)(R + 4 * DHS * 166);      // 160
  const int tid = threadIdx.x;
  const int hh = blockIdx.x & 1, b = blockIdx.x >> 1;

  for (int i = tid; i < 4 * DHS * DHS; i += 256) {
    int g = i / (DHS * DHS);
    int rem = i - g * DHS * DHS;
    int d = rem / DHS, e = rem % DHS;
    R[(g * DHS + e) * 166 + d] = f2bf(Rg[((size_t)(g * NHX + hh) * DHS + d) * DHS + e]);
  }
  if (tid < DHS) ylds[tid] = 0.f;
  __syncthreads();

  const int e = tid;
  float c = 0.f, n = 0.f, m = 0.f;
  const __bf16* r0p = R + (0 * DHS + e) * 166;
  const __bf16* r1p = R + (1 * DHS + e) * 166;
  const __bf16* r2p = R + (2 * DHS + e) * 166;
  const __bf16* r3p = R + (3 * DHS + e) * 166;

  for (int t = 0; t < SEQ; ++t) {
    float a0 = 0.f, a1 = 0.f, a2 = 0.f, a3 = 0.f;
    if (e < DHS) {
      #pragma unroll 8
      for (int d = 0; d < DHS; ++d) {
        float yd = ylds[d];
        a0 += yd * (float)r0p[d];
        a1 += yd * (float)r1p[d];
        a2 += yd * (float)r2p[d];
        a3 += yd * (float)r3p[d];
      }
    }
    __syncthreads();  // all reads of ylds complete
    if (e < DHS) {
      size_t pb = (size_t)(b * SEQ + t) * 1280 + hh * DHS + e;
      float iraw = pre[pb]        + a0;
      float fraw = pre[pb + 320]  + a1;
      float zraw = pre[pb + 640]  + a2;
      float oraw = pre[pb + 960]  + a3;
      float lfm  = m + logsigf(fraw);
      float mnew = fmaxf(iraw, lfm);
      float ii = expf(iraw - mnew), ff = expf(lfm - mnew);
      c = ff * c + ii * tanhf(zraw);
      n = ff * n + ii;
      m = mnew;
      float y = sigmf(oraw) * c / n;
      ys[(size_t)(b * SEQ + t) * EDIM + hh * DHS + e] = y;
      ylds[e] = y;
    }
    __syncthreads();
  }
}

// ================================================================ sLSTM per-head group norm * gn, += residual
__global__ __launch_bounds__(256) void k_gnorm_res(
    const float* __restrict__ ys, const float* __restrict__ gn, float* __restrict__ x)
{
  int gid = blockIdx.x * 8 + (threadIdx.x >> 5);  // token*2 + head
  int lane = threadIdx.x & 31;
  int hh = gid & 1, tok = gid >> 1;
  size_t base = (size_t)tok * EDIM + hh * DHS;
  float vals[5], s1 = 0.f, s2 = 0.f;
  #pragma unroll
  for (int i = 0; i < 5; ++i) {
    float v = ys[base + lane + 32 * i];
    vals[i] = v; s1 += v; s2 += v * v;
  }
  #pragma unroll
  for (int off = 16; off; off >>= 1) {
    s1 += __shfl_xor(s1, off, 32);
    s2 += __shfl_xor(s2, off, 32);
  }
  float mu = s1 * (1.f / DHS);
  float var = s2 * (1.f / DHS) - mu * mu;
  float rs = rsqrtf(var + 1e-5f);
  #pragma unroll
  for (int i = 0; i < 5; ++i) {
    int d = lane + 32 * i;
    x[base + d] += (vals[i] - mu) * rs * gn[hh * DHS + d];
  }
}

// ================================================================ gated FFN activation: relu(gate)*up
__global__ void k_ffn_act(const float* __restrict__ h, float* __restrict__ out)
{
  int idx = blockIdx.x * 256 + threadIdx.x;
  if (idx >= TOK * IDIM) return;
  int c = idx % IDIM, tok = idx / IDIM;
  float g = h[(size_t)tok * 832 + c];
  float u = h[(size_t)tok * 832 + IDIM + c];
  out[(size_t)tok * IDIM + c] = fmaxf(g, 0.f) * u;
}

// ================================================================ build cat = [d - pid, q_emb, fam, unf]
__global__ void k_cat(const float* __restrict__ d, const float* __restrict__ pid,
                      const float* __restrict__ qe, const int* __restrict__ rin,
                      float* __restrict__ cat)
{
  int idx = blockIdx.x * 256 + threadIdx.x;
  if (idx >= TOK * EDIM) return;
  int e = idx % EDIM, tok = idx / EDIM;
  int rr = min(max(rin[tok], 0), 1);
  float dv = d[(size_t)tok * EDIM + e];
  size_t base = (size_t)tok * 1280;
  cat[base + e] = dv - pid[tok];
  cat[base + EDIM + e] = qe[(size_t)tok * EDIM + e];
  cat[base + 2 * EDIM + e] = (rr == 1) ? dv : 0.f;
  cat[base + 3 * EDIM + e] = (rr == 0) ? dv : 0.f;
}

// ================================================================ host
extern "C" void kernel_launch(void* const* d_in, const int* in_sizes, int n_in,
                              void* d_out, int out_size, void* d_ws, size_t ws_size,
                              hipStream_t stream) {
  const int*   cin   = (const int*)d_in[0];
  const int*   rin   = (const int*)d_in[1];
  const float* qembW = (const float*)d_in[2];
  const float* qaemb = (const float*)d_in[3];
  const float* qdiff = (const float*)d_in[4];
  const float* qadif = (const float*)d_in[5];
  const float* dif   = (const float*)d_in[6];
  const float* m_ln  = (const float*)d_in[7];
  const float* pu_w  = (const float*)d_in[8];
  const float* pu_b  = (const float*)d_in[9];
  const float* cv_w  = (const float*)d_in[10];
  const float* cv_b  = (const float*)d_in[11];
  const float* q_w   = (const float*)d_in[12];
  const float* q_b   = (const float*)d_in[13];
  const float* k_w   = (const float*)d_in[14];
  const float* k_b   = (const float*)d_in[15];
  const float* v_w   = (const float*)d_in[16];
  const float* v_b   = (const float*)d_in[17];
  const float* ig_w  = (const float*)d_in[18];
  const float* ig_b  = (const float*)d_in[19];
  const float* fg_w  = (const float*)d_in[20];
  const float* fg_b  = (const float*)d_in[21];
  const float* onrm  = (const float*)d_in[22];
  const float* skip  = (const float*)d_in[23];
  const float* pd_w  = (const float*)d_in[24];
  const float* pd_b  = (const float*)d_in[25];
  const float* s_ln  = (const float*)d_in[26];
  const float* scv_w = (const float*)d_in[27];
  const float* scv_b = (const float*)d_in[28];
  const float* Wg    = (const float*)d_in[29];
  const float* Rg    = (const float*)d_in[30];
  const float* bg    = (const float*)d_in[31];
  const float* gn    = (const float*)d_in[32];
  const float* f_ln  = (const float*)d_in[33];
  const float* fu_w  = (const float*)d_in[34];
  const float* fu_b  = (const float*)d_in[35];
  const float* fd_w  = (const float*)d_in[36];
  const float* fd_b  = (const float*)d_in[37];
  const float* pnrm  = (const float*)d_in[38];
  const float* o1_w  = (const float*)d_in[39];
  const float* o1_b  = (const float*)d_in[40];
  const float* o2_w  = (const float*)d_in[41];
  const float* o2_b  = (const float*)d_in[42];
  const float* o3_w  = (const float*)d_in[43];
  const float* o3_b  = (const float*)d_in[44];

  float* ws = (float*)d_ws;
  const size_t T = TOK;
  float* QE  = ws;                    // T*320
  float* PID = QE  + T * 320;         // T
  float* X   = PID + T;               // T*320
  float* T0  = X   + T * 320;         // T*320
  float* BIG = T0  + T * 320;         // T*1280
  float* C0  = BIG + T * 1280;        // T*640
  float* C1  = C0  + T * 640;         // T*416
  float* C2  = C1  + T * 416;         // T*416
  float* C3  = C2  + T * 416;         // T*416
  float* C4  = C3  + T * 416;         // T*416
  float* IG  = C4  + T * 416;         // 32768
  float* FG  = IG  + 32768;
  float* AL  = FG  + 32768;
  float* MR  = AL  + 32768;
  float* EN  = MR  + 32768;

  auto gemm = [&](const float* A, const float* Bw, const float* bias, float* C,
                  int N, int K, int lda, int ldb, int ldc, int accum, int act) {
    dim3 g(TOK / 128, (N + 63) / 64);
    k_gemm<<<g, 256, 0, stream>>>(A, Bw, bias, C, TOK, N, K, lda, ldb, ldc, accum, act);
  };

  // ---- embeddings -> x
  k_embed<<<TOK, EDIM, 0, stream>>>(cin, rin, qembW, qaemb, qdiff, qadif, dif, X, QE, PID);

  // ---- mLSTM block
  k_ln<<<TOK / 8, 256, 0, stream>>>(X, m_ln, T0, 0);
  gemm(T0, pu_w, pu_b, BIG, 832, 320, 320, 832, 832, 0, 0);           // xi = ln_x @ proj_up
  k_conv_silu<<<(TOK * IDIM + 255) / 256, 256, 0, stream>>>(BIG, 832, cv_w, cv_b, C0, IDIM, IDIM);
  k_headwise<<<(TOK * 104 + 255) / 256, 256, 0, stream>>>(C0, IDIM, q_w, q_b, C1); // q from xc
  k_headwise<<<(TOK * 104 + 255) / 256, 256, 0, stream>>>(C0, IDIM, k_w, k_b, C2); // k from xc
  k_headwise<<<(TOK * 104 + 255) / 256, 256, 0, stream>>>(BIG, 832,  v_w, v_b, C3); // v from xm
  k_igfg<<<TOK / 8, 256, 0, stream>>>(C1, C2, C3, ig_w, ig_b, fg_w, fg_b, IG, FG);
  k_mscan<<<1, 64, 0, stream>>>(IG, FG, AL, MR, EN);
  {
    size_t smemA = (size_t)(64 * 521 + 512 + 64 + 64 + 256 + 64) * sizeof(float);
    k_attn<<<dim3(SEQ / 64, NHX, NB), 256, smemA, stream>>>(C1, C2, C3, AL, MR, EN, C4);
  }
  k_mhnorm_gate<<<TOK * NHX / 8, 256, 0, stream>>>(C4, C0, BIG, onrm, skip, C1);
  gemm(C1, pd_w, pd_b, X, 320, 416, 416, 320, 320, 1, 0);             // x += proj_down(hcomb)

  // ---- sLSTM block
  k_ln<<<TOK / 8, 256, 0, stream>>>(X, s_ln, T0, 0);
  k_conv_silu<<<(TOK * EDIM + 255) / 256, 256, 0, stream>>>(T0, 320, scv_w, scv_b, C0, 320, EDIM);
  for (int g = 0; g < 4; ++g)
    for (int h = 0; h < NHX; ++h) {
      const float* A = (g < 2 ? C0 : T0) + h * DHS;   // i,f from conv path; z,o from raw
      gemm(A, Wg + (size_t)(g * NHX + h) * DHS * DHS, bg + (g * NHX + h) * DHS,
           BIG + g * 320 + h * DHS, DHS, DHS, 320, DHS, 1280, 0, 0);
    }
  {
    size_t smemS = (size_t)4 * DHS * 166 * sizeof(__bf16) + DHS * sizeof(float);
    k_sscan<<<NB * NHX, 256, smemS, stream>>>(Rg, BIG, C3);
  }
  k_gnorm_res<<<TOK * NHX / 8, 256, 0, stream>>>(C3, gn, X);

  // ---- gated FFN
  k_ln<<<TOK / 8, 256, 0, stream>>>(X, f_ln, T0, 0);
  gemm(T0, fu_w, fu_b, BIG, 832, 320, 320, 832, 832, 0, 0);
  k_ffn_act<<<(TOK * IDIM + 255) / 256, 256, 0, stream>>>(BIG, C1);
  gemm(C1, fd_w, fd_b, X, 320, 416, 416, 320, 320, 1, 0);

  // ---- output head
  k_ln<<<TOK / 8, 256, 0, stream>>>(X, pnrm, T0, 1);                  // + nan_to_num
  k_cat<<<(TOK * EDIM + 255) / 256, 256, 0, stream>>>(T0, PID, QE, rin, BIG);
  gemm(BIG, o1_w, o1_b, C0, 640, 1280, 1280, 640, 640, 0, 1);         // relu
  gemm(C0, o2_w, o2_b, C3, 320, 640, 640, 320, 320, 0, 1);            // relu
  gemm(C3, o3_w, o3_b, (float*)d_out, 1000, 320, 320, 1000, 1000, 0, 2); // sigmoid
}
